// Attention_87058987090057
// MI455X (gfx1250) — matrix operand order repaired and checked by reading, not compiled, not run on hardware
//
#include <hip/hip_runtime.h>
#include <hip/hip_bf16.h>

// ---------------------------------------------------------------------------
// Swin-style window attention for MI455X (gfx1250), wave32 + WMMA f16.
//   B=16, N=1024 tokens, HEADS=8, HEAD_DIM=32, D_IN=D_OUT=256.
// Softmax runs in the log2 domain (log2e folded into Q-scale and bias) and
// row-sums are computed by an extra WMMA against a ones matrix.
// ---------------------------------------------------------------------------

typedef _Float16 v16h __attribute__((ext_vector_type(16)));
typedef _Float16 v8h  __attribute__((ext_vector_type(8)));
typedef float    v8f  __attribute__((ext_vector_type(8)));

#define BATCH   16
#define NTOK    1024
#define HEADS   8
#define HDIM    32
#define DMODEL  256
#define NBIAS   3969          // (2*32-1)^2 relative-position entries
#define LOG2E   1.4426950408889634f

union V16U { v16h v; v8h h[2]; };

__device__ __forceinline__ float fast_exp2(float x) {
#if __has_builtin(__builtin_amdgcn_exp2f)
    return __builtin_amdgcn_exp2f(x);      // raw v_exp_f32
#else
    return exp2f(x);
#endif
}

// WMMA: D = A(16x32 f16) * B(32x16 f16) + C(16x16 f32)
__device__ __forceinline__ v8f wmma_f16(v16h a, v16h b, v8f c) {
    return __builtin_amdgcn_wmma_f32_16x16x32_f16(
        /*neg_a=*/false, a, /*neg_b=*/false, b,
        /*c_mod=*/(short)0, c, /*reuse_a=*/false, /*reuse_b=*/false);
}

// A-matrix register layout (16-bit A 16x32), contiguous K order:
//   lanes 0-15  : row M=lane,    halves = K{0..7, 8..15}
//   lanes 16-31 : row M=lane-16, halves = K{16..23, 24..31}
__device__ __forceinline__ v16h load_a16(const _Float16* row, int hi) {
    V16U u;
    u.h[0] = *(const v8h*)(row + hi * 16);
    u.h[1] = *(const v8h*)(row + hi * 16 + 8);
    return u.v;
}

// ---------------------------------------------------------------------------
// Kernel 0: weight prep — Wq/Wk/Wv/Wo (f32, [k][n]) -> f16 transposed [n][k]
// ---------------------------------------------------------------------------
__global__ void prep_weights_kernel(const float* __restrict__ Wq,
                                    const float* __restrict__ Wk,
                                    const float* __restrict__ Wv,
                                    const float* __restrict__ Wo,
                                    _Float16* __restrict__ WTall) {
    int i = blockIdx.x * blockDim.x + threadIdx.x;      // 0 .. 4*65536-1
    if (i >= 4 * 65536) return;
    int sel = i >> 16;
    int r   = i & 65535;
    int k   = r >> 8;
    int n   = r & 255;
    const float* W = (sel == 0) ? Wq : (sel == 1) ? Wk : (sel == 2) ? Wv : Wo;
    WTall[(size_t)sel * 65536 + (size_t)n * 256 + k] = (_Float16)W[r];
}

// ---------------------------------------------------------------------------
// Kernel 1: QKV projection.  x(16384x256 f32) @ W -> f16.
//   wsel = blockIdx.y : 0=Q (scaled by log2e/sqrt(32)), 1=K, 2=V(transposed).
// ---------------------------------------------------------------------------
__global__ __launch_bounds__(256)
void qkv_kernel(const float* __restrict__ x,
                const _Float16* __restrict__ WTall,
                _Float16* __restrict__ Qf,
                _Float16* __restrict__ Kf,
                _Float16* __restrict__ Vt) {
    const int wsel = blockIdx.y;
    const _Float16* WT = WTall + (size_t)wsel * 65536;

    const int tid = threadIdx.x;
    const int wave = tid >> 5, lane = tid & 31;
    const int laneRow = lane & 15, hi = lane >> 4;
    const int m0 = blockIdx.x * 128 + wave * 16;        // 16 rows per wave

    // Preload A registers: 16 rows x 256 K-dims (8 k-tiles), f32 -> f16.
    v16h areg[8];
    const float* xr = x + (size_t)(m0 + laneRow) * DMODEL;
#pragma unroll
    for (int kt = 0; kt < 8; ++kt) {
#pragma unroll
        for (int j = 0; j < 8; ++j) {
            areg[kt][j]     = (_Float16)xr[kt * 32 + hi * 16 + j];
            areg[kt][8 + j] = (_Float16)xr[kt * 32 + hi * 16 + 8 + j];
        }
    }

    // Q carries both the 1/sqrt(d) scale and log2e for exp2-domain softmax.
    const float scale = (wsel == 0) ? (0.17677669529663687f * LOG2E) : 1.0f;
    const _Float16* wrow = WT + (size_t)laneRow * 256 + hi * 16;

    v16h bbuf[2][8];
#pragma unroll
    for (int kt = 0; kt < 8; ++kt)
        bbuf[0][kt] = *(const v16h*)(wrow + kt * 32);

#pragma unroll
    for (int nt = 0; nt < 16; ++nt) {
        const int n0  = nt * 16;
        const int cur = nt & 1;
        v8f acc = {};
#pragma unroll
        for (int kt = 0; kt < 8; ++kt) {
            if (nt < 15)   // prefetch next nt's B tile while WMMAs execute
                bbuf[cur ^ 1][kt] =
                    *(const v16h*)(wrow + (size_t)(nt + 1) * 16 * 256 + kt * 32);
            acc = wmma_f16(areg[kt], bbuf[cur][kt], acc);
        }
        const int c  = n0 + laneRow;                    // output column 0..255
        const int hh = c >> 5, d = c & 31;
#pragma unroll
        for (int v = 0; v < 8; ++v) {
            const int m  = m0 + v + 8 * hi;
            const int bb = m >> 10, n = m & 1023;
            const size_t bh = (size_t)(bb * HEADS + hh);
            const _Float16 val = (_Float16)(acc[v] * scale);
            if (wsel == 0)      Qf[bh * 32768 + (size_t)n * 32 + d] = val;
            else if (wsel == 1) Kf[bh * 32768 + (size_t)n * 32 + d] = val;
            else                Vt[bh * 32768 + (size_t)d * 1024 + n] = val;
        }
    }
}

// ---------------------------------------------------------------------------
// Kernel 2: fused flash attention per (b, h, 128-query block).
//   - bias_table[:,h] staged into LDS via gfx1250 async-to-LDS loads,
//     then scaled by log2e in place
//   - K/V tiles for chunk t+1 stream while chunk t's softmax runs
//   - P bounced through per-wave LDS scratch (C-layout -> A-layout)
//   - row-sums accumulated by WMMA against a ones matrix (no sum shuffles)
// ---------------------------------------------------------------------------
__global__ __launch_bounds__(256)
void attn_kernel(const _Float16* __restrict__ Qf,
                 const _Float16* __restrict__ Kf,
                 const _Float16* __restrict__ Vt,
                 const float* __restrict__ bias_table,
                 _Float16* __restrict__ Att) {
    __shared__ float    biasS[NBIAS];                   // 15.9 KB: log2e*bias[:,h]
    __shared__ _Float16 Ps[8 * 512];                    // per-wave 16x32 P tile

    const int tid = threadIdx.x;
    const int wave = tid >> 5, lane = tid & 31;
    const int laneRow = lane & 15, hi = lane >> 4;

    const int bid  = blockIdx.x;
    const int qblk = bid & 7;
    const int h    = (bid >> 3) & 7;
    const int b    = bid >> 6;
    const int q0   = qblk * 128 + wave * 16;

    // ---- stage bias column h into LDS with async copy (ASYNCcnt-tracked)
    for (int i = tid; i < NBIAS; i += 256) {
        unsigned lds_addr = (unsigned)(size_t)&biasS[i];
        unsigned goff     = (unsigned)((i * HEADS + h) * 4);
        asm volatile("global_load_async_to_lds_b32 %0, %1, %2"
                     :: "v"(lds_addr), "v"(goff), "s"(bias_table)
                     : "memory");
    }
    asm volatile("s_wait_asynccnt 0x0" ::: "memory");
    __syncthreads();
    // scale to log2 domain in place
    for (int i = tid; i < NBIAS; i += 256) biasS[i] *= LOG2E;
    __syncthreads();

    const size_t bh = (size_t)(b * HEADS + h);
    const _Float16* qb = Qf + bh * (NTOK * HDIM);
    const _Float16* kb = Kf + bh * (NTOK * HDIM);
    const _Float16* vb = Vt + bh * (HDIM * NTOK);

    const v16h qa = load_a16(qb + (size_t)(q0 + laneRow) * HDIM, hi);

    v16h ones;
#pragma unroll
    for (int j = 0; j < 16; ++j) ones[j] = (_Float16)1.0f;

    float rmax[8];
    int   qlin[8];
    v8f acc0 = {}, acc1 = {}, rs = {};
#pragma unroll
    for (int v = 0; v < 8; ++v) {
        rmax[v] = -3.0e38f;
        const int q = q0 + v + 8 * hi;
        qlin[v] = (q >> 5) * 63 + (q & 31) + 1984;      // + (31*63 + 31)
    }

    _Float16* pw = Ps + wave * 512;

    // chunk-tile loaders (keys n0..n0+15 / n0+16..n0+31; V transposed [d][n])
    const _Float16* krow0 = kb + (size_t)laneRow * HDIM + hi * 16;
    const _Float16* vrow0 = vb + (size_t)laneRow * NTOK + hi * 16;
    const _Float16* vrow1 = vb + (size_t)(16 + laneRow) * NTOK + hi * 16;

    // preload chunk 0
    v16h kb0 = *(const v16h*)(krow0);
    v16h kb1 = *(const v16h*)(krow0 + 16 * HDIM);
    v16h vb0 = *(const v16h*)(vrow0);
    v16h vb1 = *(const v16h*)(vrow1);

    for (int kt = 0; kt < NTOK / 32; ++kt) {
        const int n0 = kt * 32;
        // ---- scores (log2 domain): Q(16x32) @ K^T tiles
        v8f s0 = {}, s1 = {};
        s0 = wmma_f16(qa, kb0, s0);
        s1 = wmma_f16(qa, kb1, s1);

        // ---- prefetch next chunk's K/V tiles (in flight during softmax)
        const int nn0 = ((kt + 1) & 31) * 32;
        const v16h nk0 = *(const v16h*)(krow0 + (size_t)nn0 * HDIM);
        const v16h nk1 = *(const v16h*)(krow0 + (size_t)(nn0 + 16) * HDIM);
        const v16h nv0 = *(const v16h*)(vrow0 + nn0);
        const v16h nv1 = *(const v16h*)(vrow1 + nn0);

        // ---- relative-position bias (LDS) + online softmax (base-2)
        const int k0c  = n0 + laneRow;
        const int k1c  = k0c + 16;
        const int klin0 = (k0c >> 5) * 63 + (k0c & 31);
        const int klin1 = (k1c >> 5) * 63 + (k1c & 31);
#pragma unroll
        for (int v = 0; v < 8; ++v) {
            const float e0 = s0[v] + biasS[qlin[v] - klin0];
            const float e1 = s1[v] + biasS[qlin[v] - klin1];

            float m = fmaxf(e0, e1);
            m = fmaxf(m, __shfl_xor(m, 1, 32));
            m = fmaxf(m, __shfl_xor(m, 2, 32));
            m = fmaxf(m, __shfl_xor(m, 4, 32));
            m = fmaxf(m, __shfl_xor(m, 8, 32));

            const float nm   = fmaxf(rmax[v], m);
            const float corr = fast_exp2(rmax[v] - nm);
            rmax[v] = nm;

            const float p0 = fast_exp2(e0 - nm);
            const float p1 = fast_exp2(e1 - nm);
            acc0[v] = acc0[v] * corr;
            acc1[v] = acc1[v] * corr;
            rs[v]   = rs[v]   * corr;

            const int prow = v + 8 * hi;
            pw[prow * 32 + laneRow]      = (_Float16)p0;
            pw[prow * 32 + 16 + laneRow] = (_Float16)p1;
        }

        // ---- reload P in A-layout (same-wave LDS ordering)
        const v16h pa = load_a16(pw + laneRow * 32, hi);

        // ---- out += P @ V ; row-sums += P @ ones (lands in matching slots)
        acc0 = wmma_f16(pa, vb0, acc0);
        acc1 = wmma_f16(pa, vb1, acc1);
        rs   = wmma_f16(pa, ones, rs);

        kb0 = nk0; kb1 = nk1; vb0 = nv0; vb1 = nv1;
    }

    // ---- normalize and store Att[b, n, h*32 + d] (f16)
    const size_t obase = (size_t)b * NTOK * DMODEL;
#pragma unroll
    for (int v = 0; v < 8; ++v) {
        const int q = q0 + v + 8 * hi;
        const float inv = 1.0f / rs[v];
        Att[obase + (size_t)q * DMODEL + h * 32 + laneRow]      = (_Float16)(acc0[v] * inv);
        Att[obase + (size_t)q * DMODEL + h * 32 + 16 + laneRow] = (_Float16)(acc1[v] * inv);
    }
}

// ---------------------------------------------------------------------------
// Kernel 3: output projection.  Att(16384x256 f16) @ Wo + bo -> f32 out.
// ---------------------------------------------------------------------------
__global__ __launch_bounds__(256)
void proj_kernel(const _Float16* __restrict__ Att,
                 const _Float16* __restrict__ WoT,
                 const float* __restrict__ bo,
                 float* __restrict__ out) {
    const int tid = threadIdx.x;
    const int wave = tid >> 5, lane = tid & 31;
    const int laneRow = lane & 15, hi = lane >> 4;
    const int m0 = blockIdx.x * 128 + wave * 16;

    v16h areg[8];
    const _Float16* ar = Att + (size_t)(m0 + laneRow) * DMODEL;
#pragma unroll
    for (int kt = 0; kt < 8; ++kt) areg[kt] = load_a16(ar + kt * 32, hi);

    const _Float16* wrow = WoT + (size_t)laneRow * 256 + hi * 16;

    v16h bbuf[2][8];
#pragma unroll
    for (int kt = 0; kt < 8; ++kt)
        bbuf[0][kt] = *(const v16h*)(wrow + kt * 32);

#pragma unroll
    for (int nt = 0; nt < 16; ++nt) {
        const int n0  = nt * 16;
        const int cur = nt & 1;
        v8f acc = {};
#pragma unroll
        for (int kt = 0; kt < 8; ++kt) {
            if (nt < 15)
                bbuf[cur ^ 1][kt] =
                    *(const v16h*)(wrow + (size_t)(nt + 1) * 16 * 256 + kt * 32);
            acc = wmma_f16(areg[kt], bbuf[cur][kt], acc);
        }
        const float bias = bo[n0 + laneRow];
#pragma unroll
        for (int v = 0; v < 8; ++v) {
            const int m = m0 + v + 8 * hi;
            out[(size_t)m * DMODEL + n0 + laneRow] = acc[v] + bias;
        }
    }
}

// ---------------------------------------------------------------------------
// Launch: inputs = {x, Wq, Wk, Wv, bias_table, Wo, bo}
// Workspace (f16 elems): WT[4*65536] | Q[4M] | K[4M] | Vt[4M] | Att[4M]
// ---------------------------------------------------------------------------
extern "C" void kernel_launch(void* const* d_in, const int* in_sizes, int n_in,
                              void* d_out, int out_size, void* d_ws, size_t ws_size,
                              hipStream_t stream) {
    (void)in_sizes; (void)n_in; (void)out_size; (void)ws_size;

    const float* x          = (const float*)d_in[0];
    const float* Wq         = (const float*)d_in[1];
    const float* Wk         = (const float*)d_in[2];
    const float* Wv         = (const float*)d_in[3];
    const float* bias_table = (const float*)d_in[4];
    const float* Wo         = (const float*)d_in[5];
    const float* bo         = (const float*)d_in[6];
    float*       out        = (float*)d_out;

    _Float16* ws  = (_Float16*)d_ws;
    _Float16* WT  = ws;                                  // 4 * 65536
    _Float16* Qf  = ws + 4 * 65536;                      // 16*8*1024*32
    _Float16* Kf  = Qf + (size_t)BATCH * HEADS * NTOK * HDIM;
    _Float16* Vt  = Kf + (size_t)BATCH * HEADS * NTOK * HDIM;
    _Float16* Att = Vt + (size_t)BATCH * HEADS * NTOK * HDIM;

    prep_weights_kernel<<<(4 * 65536 + 255) / 256, 256, 0, stream>>>(
        Wq, Wk, Wv, Wo, WT);

    qkv_kernel<<<dim3(128, 3), 256, 0, stream>>>(x, WT, Qf, Kf, Vt);

    attn_kernel<<<BATCH * HEADS * (NTOK / 128), 256, 0, stream>>>(
        Qf, Kf, Vt, bias_table, Att);

    proj_kernel<<<128, 256, 0, stream>>>(Att, WT + 3 * 65536, bo, out);
}
